// DynamicMultiHeadAttention_23811298689632
// MI455X (gfx1250) — matrix operand order repaired and checked by reading, not compiled, not run on hardware
//
#include <hip/hip_runtime.h>
#include <hip/hip_bf16.h>
#include <stdint.h>

// ---------------------------------------------------------------------------
// MI455X (gfx1250) multi-head attention, flash-style, bf16 WMMA everywhere,
// async GLOBAL->LDS staging (ASYNCcnt) where layouts permit.
// B=4, S=2048, D=1024, H=16, HD=64. wave32; WMMA 16x16x32 bf16->f32.
// ---------------------------------------------------------------------------

#define Bn 4
#define Sn 2048
#define Dn 1024
#define Hn 16
#define HDn 64
#define Mrows (Bn * Sn)   // 8192

typedef __attribute__((ext_vector_type(16))) __bf16 v16bf;
typedef __attribute__((ext_vector_type(8)))  float  v8f;

#define AS1 __attribute__((address_space(1)))
#define AS3 __attribute__((address_space(3)))

// exact parameter type of the async-LDS builtins (from hipcc diagnostic):
// int __vector(4), AS1 for the global side, AS3 for the LDS side.
typedef int vi4 __attribute__((vector_size(4 * sizeof(int))));

#if defined(__has_builtin)
#if __has_builtin(__builtin_amdgcn_global_load_async_to_lds_b128)
#define HAVE_ASYNC_LDS 1
#endif
#endif

// 16B global -> LDS copy. Async (ASYNCcnt) when the gfx1250 builtin exists.
static __device__ __forceinline__ void async_cp16(const __bf16* g, __bf16* l) {
#ifdef HAVE_ASYNC_LDS
  __builtin_amdgcn_global_load_async_to_lds_b128(
      (AS1 vi4*)(uintptr_t)g,
      (AS3 vi4*)(uint32_t)(uintptr_t)l,   // low 32 bits of flat = LDS offset
      /*offset=*/0, /*cpol=*/0);
#else
  *(uint4*)l = *(const uint4*)g;
#endif
}
static __device__ __forceinline__ void wait_async() {
#ifdef HAVE_ASYNC_LDS
#if __has_builtin(__builtin_amdgcn_s_wait_asynccnt)
  __builtin_amdgcn_s_wait_asynccnt(0);
#else
  asm volatile("s_wait_asynccnt 0x0" ::: "memory");
#endif
#endif
}

static __device__ __forceinline__ __bf16 f2bf(float f) { return (__bf16)f; }

// ---- WMMA fragment loaders (CDNA5 16-bit layouts, ISA 7.12.2) --------------
// A (16x32, MxK): lanes 0-15 row M=lane, K={0..7,16..23}; lanes 16-31 same
// row, K={8..15,24..31}.
static __device__ __forceinline__ v16bf load_a_frag(const __bf16* base,
                                                    int stride, int lane) {
  const int hf = (lane >> 4) & 1, m = lane & 15;
  const __bf16* p = base + m * stride + hf * 8;
  v16bf f;
#pragma unroll
  for (int e = 0; e < 8; ++e) { f[e] = p[e]; f[8 + e] = p[16 + e]; }
  return f;
}
// B (32x16, KxN): lanes 0-15 col N=lane K={0..7,16..23}; lanes 16-31 col N K={8..15,24..31}.
// `base` = column-major tile (contraction contiguous per column, pitch cstride)
static __device__ __forceinline__ v16bf load_b_frag(const __bf16* base,
                                                    int cstride, int lane) {
  const int hf = (lane >> 4) & 1, n = lane & 15;
  const __bf16* p = base + n * cstride + hf * 8;
  v16bf f;
#pragma unroll
  for (int e = 0; e < 8; ++e) { f[e] = p[e]; f[8 + e] = p[16 + e]; }
  return f;
}

static __device__ __forceinline__ v8f wmma_bf16(v16bf a, v16bf b, v8f c) {
  return __builtin_amdgcn_wmma_f32_16x16x32_bf16(false, a, false, b,
                                                 (short)0, c, false, false);
}

// reductions within a 16-lane half (xor masks < 16 stay in the half)
static __device__ __forceinline__ float hred_max(float v) {
#pragma unroll
  for (int m = 1; m <= 8; m <<= 1) v = fmaxf(v, __shfl_xor(v, m, 32));
  return v;
}
static __device__ __forceinline__ float hred_sum(float v) {
#pragma unroll
  for (int m = 1; m <= 8; m <<= 1) v += __shfl_xor(v, m, 32);
  return v;
}

// ---------------------------------------------------------------------------
// Kernel 1: f32 weight [k][n] -> bf16 transposed [n][k] (so LDS W tiles are
// row-contiguous and eligible for async B128 staging).
// ---------------------------------------------------------------------------
__global__ void cvt_wT_kernel(const float* __restrict__ w,
                              __bf16* __restrict__ wT) {
  int idx = blockIdx.x * 256 + threadIdx.x;   // 1M elements
  int k = idx >> 10, n = idx & 1023;
  wT[(size_t)n * Dn + k] = f2bf(w[idx]);
}

// ---------------------------------------------------------------------------
// Kernel 2: fused QKV projection. grid=(M/64, D/128, 3), block=256 (8 waves).
// C tile 64x128; wave w -> rows (w>>1)*16, cols (w&1)*64 (4 accs). K-step 64.
// q output pre-scaled by 1/sqrt(64)=0.125 after bias.
// ---------------------------------------------------------------------------
__global__ __launch_bounds__(256) void qkv_proj_kernel(
    const float* __restrict__ x,
    const float* __restrict__ peq, const float* __restrict__ pek,
    const float* __restrict__ pev,
    const __bf16* __restrict__ wqT, const __bf16* __restrict__ wkT,
    const __bf16* __restrict__ wvT,
    const float* __restrict__ bq, const float* __restrict__ bk,
    const float* __restrict__ bv,
    __bf16* __restrict__ qo, __bf16* __restrict__ ko,
    __bf16* __restrict__ vo) {
  __shared__ __align__(16) __bf16 lds_a[64 * 64];    // [r][k]
  __shared__ __align__(16) __bf16 lds_w[128 * 64];   // [n][k] (Wt rows)

  const int z = blockIdx.z;
  const float*  pe   = (z == 0) ? peq : (z == 1) ? pek : pev;
  const __bf16* wT   = (z == 0) ? wqT : (z == 1) ? wkT : wvT;
  const float*  bias = (z == 0) ? bq  : (z == 1) ? bk  : bv;
  __bf16*       out  = (z == 0) ? qo  : (z == 1) ? ko  : vo;
  const float oscale = (z == 0) ? 0.125f : 1.0f;

  const int tid = threadIdx.x, lane = tid & 31, wid = tid >> 5;
  const int rg = wid >> 1, cg = wid & 1;
  const int row0 = blockIdx.x * 64, col0 = blockIdx.y * 128;

  v8f acc[4] = {};
  for (int kk = 0; kk < Dn; kk += 64) {
    // W tile: 128 rows x 128B -> async B128 (1024 chunks / 256 thr = 4 each)
    for (int ci = tid; ci < 128 * 8; ci += 256) {
      int r = ci >> 3, p = ci & 7;
      async_cp16(wT + (size_t)(col0 + r) * Dn + kk + p * 8,
                 lds_w + r * 64 + p * 8);
    }
    // A tile: (x + pos_emb) f32 -> bf16 (must pass through VALU)
    for (int idx = tid; idx < 64 * 64; idx += 256) {
      int r = idx >> 6, c = idx & 63;
      int grow = row0 + r, s = grow & (Sn - 1);
      lds_a[r * 64 + c] =
          f2bf(x[(size_t)grow * Dn + kk + c] + pe[(size_t)s * Dn + kk + c]);
    }
    wait_async();
    __syncthreads();
    v16bf a0 = load_a_frag(lds_a + rg * 16 * 64, 64, lane);
    v16bf a1 = load_a_frag(lds_a + rg * 16 * 64 + 32, 64, lane);
#pragma unroll
    for (int t = 0; t < 4; ++t) {
      const __bf16* wb = lds_w + (cg * 64 + t * 16) * 64;
      acc[t] = wmma_bf16(a0, load_b_frag(wb, 64, lane), acc[t]);
      acc[t] = wmma_bf16(a1, load_b_frag(wb + 32, 64, lane), acc[t]);
    }
    __syncthreads();
  }
  const int hf = lane >> 4, n = lane & 15;
  const int grow0 = row0 + rg * 16;
#pragma unroll
  for (int t = 0; t < 4; ++t)
#pragma unroll
    for (int i = 0; i < 8; ++i) {
      int m = i + 8 * hf;
      int c = col0 + cg * 64 + t * 16 + n;
      out[(size_t)(grow0 + m) * Dn + c] = f2bf((acc[t][i] + bias[c]) * oscale);
    }
}

// ---------------------------------------------------------------------------
// Kernel 3: flash attention. grid=(S/64, H, B), block=128 (4 waves).
// Wave = 16 q-rows x full head; keys streamed in 64-wide tiles (16 WMMA/iter).
// ---------------------------------------------------------------------------
__global__ __launch_bounds__(128) void flash_attn_kernel(
    const __bf16* __restrict__ Q, const __bf16* __restrict__ K,
    const __bf16* __restrict__ V,
    const float* __restrict__ bias2d,   // [S,S]
    const float* __restrict__ bias_kv,  // [B,S]
    __bf16* __restrict__ X) {
  __shared__ __align__(16) __bf16 k_lds[64 * 64];     // [key][hd]
  __shared__ __align__(16) __bf16 v_lds[64 * 64];     // [hd][key] (transposed)
  __shared__ __align__(16) __bf16 p_lds[4][16 * 64];  // per-wave P staging

  const int tid = threadIdx.x, lane = tid & 31, wid = tid >> 5;
  const int hf = lane >> 4, n = lane & 15;
  const int b = blockIdx.z, h = blockIdx.y;
  const int q0 = blockIdx.x * 64 + wid * 16;
  const size_t qbase = ((size_t)b * Sn + q0) * Dn + h * HDn;

  v16bf aq0 = load_a_frag(Q + qbase, Dn, lane);
  v16bf aq1 = load_a_frag(Q + qbase + 32, Dn, lane);

  v8f o[4] = {};
  float mrow[8], lrow[8];
#pragma unroll
  for (int i = 0; i < 8; ++i) { mrow[i] = -1e30f; lrow[i] = 0.0f; }

  for (int kt = 0; kt < Sn; kt += 64) {
    // K tile: async B128 (512 chunks / 128 thr = 4 each)
    for (int ci = tid; ci < 64 * 8; ci += 128) {
      int r = ci >> 3, p = ci & 7;
      async_cp16(K + ((size_t)b * Sn + kt + r) * Dn + h * HDn + p * 8,
                 k_lds + r * 64 + p * 8);
    }
    // V tile transposed (element transpose must go through VGPRs)
    for (int idx = tid; idx < 64 * 64; idx += 128) {
      int kr = idx >> 6, hd = idx & 63;
      v_lds[hd * 64 + kr] = V[((size_t)b * Sn + kt + kr) * Dn + h * HDn + hd];
    }
    __builtin_prefetch(&bias2d[(size_t)q0 * Sn + kt], 0, 1);
    wait_async();
    __syncthreads();

    // scores: 4 key sub-tiles x (2 hd chunks) = 8 WMMA
    v8f s[4] = {};
#pragma unroll
    for (int j = 0; j < 4; ++j) {
      const __bf16* kb = k_lds + (j * 16) * 64;
      s[j] = wmma_bf16(aq0, load_b_frag(kb, 64, lane), s[j]);
      s[j] = wmma_bf16(aq1, load_b_frag(kb + 32, 64, lane), s[j]);
    }

    float kvb[4];
#pragma unroll
    for (int j = 0; j < 4; ++j) kvb[j] = bias_kv[b * Sn + kt + j * 16 + n];

    // bias add + online softmax (row m = i + 8*hf, col n)
#pragma unroll
    for (int i = 0; i < 8; ++i) {
      int m = i + 8 * hf, qs = q0 + m;
      float v[4];
#pragma unroll
      for (int j = 0; j < 4; ++j)
        v[j] = s[j][i] + bias2d[(size_t)qs * Sn + kt + j * 16 + n] + kvb[j];
      float tmax = fmaxf(fmaxf(v[0], v[1]), fmaxf(v[2], v[3]));
      tmax = hred_max(tmax);
      float mnew = fmaxf(mrow[i], tmax);
      float corr = __expf(mrow[i] - mnew);
      mrow[i] = mnew;
      float psum = 0.0f;
#pragma unroll
      for (int j = 0; j < 4; ++j) {
        float e = __expf(v[j] - mnew);
        psum += e;
        p_lds[wid][m * 64 + j * 16 + n] = f2bf(e);
      }
      lrow[i] = lrow[i] * corr + hred_sum(psum);
#pragma unroll
      for (int t = 0; t < 4; ++t) o[t][i] *= corr;
    }
    // wave-local LDS visibility (per-wave staging, lockstep wave32)
    asm volatile("s_wait_dscnt 0x0" ::: "memory");
    v16bf ap0 = load_a_frag(&p_lds[wid][0], 64, lane);
    v16bf ap1 = load_a_frag(&p_lds[wid][0] + 32, 64, lane);

    // O += P(16x64) x V(64x64): 4 col tiles x 2 key chunks = 8 WMMA
#pragma unroll
    for (int t = 0; t < 4; ++t) {
      const __bf16* vb = v_lds + (t * 16) * 64;
      o[t] = wmma_bf16(ap0, load_b_frag(vb, 64, lane), o[t]);
      o[t] = wmma_bf16(ap1, load_b_frag(vb + 32, 64, lane), o[t]);
    }
    __syncthreads();
  }

  // normalize and emit X[b, s, h*HD + hd] as bf16
#pragma unroll
  for (int i = 0; i < 8; ++i) {
    int m = i + 8 * hf;
    float inv = 1.0f / lrow[i];
    size_t base = ((size_t)b * Sn + q0 + m) * Dn + h * HDn;
#pragma unroll
    for (int t = 0; t < 4; ++t) X[base + t * 16 + n] = f2bf(o[t][i] * inv);
  }
}

// ---------------------------------------------------------------------------
// Kernel 4: output projection X[8192,1024] x Wo[1024,1024] + bo -> f32.
// Both tiles staged with async B128 (X already bf16 row-major).
// ---------------------------------------------------------------------------
__global__ __launch_bounds__(256) void out_proj_kernel(
    const __bf16* __restrict__ X, const __bf16* __restrict__ woT,
    const float* __restrict__ bo, float* __restrict__ out) {
  __shared__ __align__(16) __bf16 lds_a[64 * 64];
  __shared__ __align__(16) __bf16 lds_w[128 * 64];

  const int tid = threadIdx.x, lane = tid & 31, wid = tid >> 5;
  const int rg = wid >> 1, cg = wid & 1;
  const int row0 = blockIdx.x * 64, col0 = blockIdx.y * 128;

  v8f acc[4] = {};
  for (int kk = 0; kk < Dn; kk += 64) {
    for (int ci = tid; ci < 64 * 8; ci += 256) {
      int r = ci >> 3, p = ci & 7;
      async_cp16(X + (size_t)(row0 + r) * Dn + kk + p * 8,
                 lds_a + r * 64 + p * 8);
    }
    for (int ci = tid; ci < 128 * 8; ci += 256) {
      int r = ci >> 3, p = ci & 7;
      async_cp16(woT + (size_t)(col0 + r) * Dn + kk + p * 8,
                 lds_w + r * 64 + p * 8);
    }
    wait_async();
    __syncthreads();
    v16bf a0 = load_a_frag(lds_a + rg * 16 * 64, 64, lane);
    v16bf a1 = load_a_frag(lds_a + rg * 16 * 64 + 32, 64, lane);
#pragma unroll
    for (int t = 0; t < 4; ++t) {
      const __bf16* wb = lds_w + (cg * 64 + t * 16) * 64;
      acc[t] = wmma_bf16(a0, load_b_frag(wb, 64, lane), acc[t]);
      acc[t] = wmma_bf16(a1, load_b_frag(wb + 32, 64, lane), acc[t]);
    }
    __syncthreads();
  }
  const int hf = lane >> 4, n = lane & 15;
  const int grow0 = row0 + rg * 16;
#pragma unroll
  for (int t = 0; t < 4; ++t)
#pragma unroll
    for (int i = 0; i < 8; ++i) {
      int m = i + 8 * hf;
      int c = col0 + cg * 64 + t * 16 + n;
      out[(size_t)(grow0 + m) * Dn + c] = acc[t][i] + bo[c];
    }
}

// ---------------------------------------------------------------------------
// Launch. Workspace (>= 72 MiB):
//   [0,2M) wqT  [2M,4M) wkT  [4M,6M) wvT  [6M,8M) woT   (bf16, transposed)
//   [8M,24M) Q  [24M,40M) K  [40M,56M) V  [56M,72M) X   (bf16, row-major)
// ---------------------------------------------------------------------------
extern "C" void kernel_launch(void* const* d_in, const int* in_sizes, int n_in,
                              void* d_out, int out_size, void* d_ws,
                              size_t ws_size, hipStream_t stream) {
  (void)in_sizes; (void)n_in; (void)out_size; (void)ws_size;
  const float* x       = (const float*)d_in[0];
  const float* peq     = (const float*)d_in[1];
  const float* pek     = (const float*)d_in[2];
  const float* pev     = (const float*)d_in[3];
  const float* bias2d  = (const float*)d_in[4];
  const float* bias_kv = (const float*)d_in[5];
  const float* wq      = (const float*)d_in[6];
  const float* bq      = (const float*)d_in[7];
  const float* wk      = (const float*)d_in[8];
  const float* bk      = (const float*)d_in[9];
  const float* wv      = (const float*)d_in[10];
  const float* bv      = (const float*)d_in[11];
  const float* wo      = (const float*)d_in[12];
  const float* bo      = (const float*)d_in[13];
  float* out = (float*)d_out;

  char* ws = (char*)d_ws;
  const size_t MB = 1u << 20;
  __bf16* wqT = (__bf16*)(ws + 0 * MB);
  __bf16* wkT = (__bf16*)(ws + 2 * MB);
  __bf16* wvT = (__bf16*)(ws + 4 * MB);
  __bf16* woT = (__bf16*)(ws + 6 * MB);
  __bf16* Qb  = (__bf16*)(ws + 8 * MB);
  __bf16* Kb  = (__bf16*)(ws + 24 * MB);
  __bf16* Vb  = (__bf16*)(ws + 40 * MB);
  __bf16* Xb  = (__bf16*)(ws + 56 * MB);

  const int WN = Dn * Dn;
  cvt_wT_kernel<<<WN / 256, 256, 0, stream>>>(wq, wqT);
  cvt_wT_kernel<<<WN / 256, 256, 0, stream>>>(wk, wkT);
  cvt_wT_kernel<<<WN / 256, 256, 0, stream>>>(wv, wvT);
  cvt_wT_kernel<<<WN / 256, 256, 0, stream>>>(wo, woT);

  qkv_proj_kernel<<<dim3(Mrows / 64, Dn / 128, 3), 256, 0, stream>>>(
      x, peq, pek, pev, wqT, wkT, wvT, bq, bk, bv, Qb, Kb, Vb);

  flash_attn_kernel<<<dim3(Sn / 64, Hn, Bn), 128, 0, stream>>>(
      Qb, Kb, Vb, bias2d, bias_kv, Xb);

  out_proj_kernel<<<dim3(Mrows / 64, Dn / 128, 1), 256, 0, stream>>>(
      Xb, woT, bo, out);
}
